// SeqLabelGenerator_69801808495263
// MI455X (gfx1250) — matrix operand order repaired and checked by reading, not compiled, not run on hardware
//
#include <hip/hip_runtime.h>
#include <hip/hip_bf16.h>

// ---------------------------------------------------------------------------
// SeqLabelGenerator for MI455X (gfx1250)
//   - bf16 WMMA (v_wmma_f32_16x16x32_bf16) for all GEMM stages
//   - fragment-major packed B matrices (contiguous 32B per lane per fragment)
//   - fragment-major bf16 preact hand-off between GEMM and LSTM recurrence
//   - async global->LDS staging of A tiles (ASYNCcnt) where available
// ---------------------------------------------------------------------------

typedef __bf16 bf16_t;
typedef __attribute__((ext_vector_type(16))) __bf16 v16bf;
typedef __attribute__((ext_vector_type(8)))  __bf16 v8bf;
typedef __attribute__((ext_vector_type(8)))  float  v8f;

#define B_SZ 32
#define L_SZ 512
#define D_SZ 192
#define H_SZ 256
#define G4   1024   // 4*H, also padded decoder N; always 64 n-tiles
#define NTGT 1000
#define NTP  1024
#define CHL  16

// ---------------------------------------------------------------------------
// Async global->LDS copy (16B per lane), guarded so compile never breaks.
// Builtin signature (from toolchain diagnostic): takes v4i32 pointers,
// (global src, lds dst, imm offset, imm cpol).
// ---------------------------------------------------------------------------
#if defined(__has_builtin)
#if __has_builtin(__builtin_amdgcn_global_load_async_to_lds_b128)
#define HAVE_ASYNC_LDS 1
#endif
#if __has_builtin(__builtin_amdgcn_s_wait_asynccnt)
#define HAVE_ASYNC_WAIT 1
#endif
#endif

typedef int v4i_vs __attribute__((__vector_size__(16)));
typedef __attribute__((address_space(1))) v4i_vs* gv4i_ptr;
typedef __attribute__((address_space(3))) v4i_vs* lv4i_ptr;

__device__ inline void async_copy_b128(void* lds_dst, const void* gsrc) {
#ifdef HAVE_ASYNC_LDS
  __builtin_amdgcn_global_load_async_to_lds_b128((gv4i_ptr)gsrc, (lv4i_ptr)lds_dst, 0, 0);
#else
  *(float4*)lds_dst = *(const float4*)gsrc;
#endif
}

__device__ inline void async_join() {
#ifdef HAVE_ASYNC_WAIT
  __builtin_amdgcn_s_wait_asynccnt(0);
#endif
  __syncthreads();
}

// ---------------------------------------------------------------------------
// WMMA fragment helpers
// ---------------------------------------------------------------------------

// A fragment (16x32 bf16) from row-major source (global or LDS), lda in elems.
// Two contiguous 16B runs per lane -> b128 loads.
__device__ inline v16bf load_a_frag(const bf16_t* A, int lda, int row0, int kbase, int lane) {
  v16bf a;
  int m  = row0 + (lane & 15);
  int kh = (lane >> 4) << 3;
  const bf16_t* p = A + (size_t)m * lda + kbase + kh;
#pragma unroll
  for (int e = 0; e < 8; ++e) a[e] = p[e];
#pragma unroll
  for (int e = 0; e < 8; ++e) a[8 + e] = p[16 + e];
  return a;
}

// Packed B fragment: Bp laid out [kc][nt(64)][lane(32)][e(16)] contiguous.
__device__ inline v16bf load_b_packed(const bf16_t* Bp, int kc, int nt, int lane) {
  return *(const v16bf*)(Bp + ((((size_t)kc << 6) + nt) * 32 + lane) * 16);
}

__device__ inline v8f wmma_bf16(v16bf a, v16bf b, v8f c) {
  return __builtin_amdgcn_wmma_f32_16x16x32_bf16(false, a, false, b, (short)0, c, false, false);
}

// ---------------------------------------------------------------------------
// Prep kernels
// ---------------------------------------------------------------------------

// Pack f32 [K, Nsrc] weight into fragment-major bf16 with N padded to 1024.
// dst[i]: e=i&15, lane=(i>>4)&31, frag=i>>9, kc=frag>>6, nt=frag&63
// k = kc*32 + ((lane>>4)<<3) + (e&7) + ((e>>3)<<4), n = nt*16 + (lane&15)
__global__ void pack_b_kernel(const float* __restrict__ src, bf16_t* __restrict__ dst,
                              int total, int Nsrc) {
  int i = blockIdx.x * 256 + threadIdx.x;
  if (i >= total) return;
  int e    = i & 15;
  int lane = (i >> 4) & 31;
  int frag = i >> 9;
  int kc = frag >> 6, nt = frag & 63;
  int k = kc * 32 + ((lane >> 4) << 3) + (e & 7) + ((e >> 3) << 4);
  int n = nt * 16 + (lane & 15);
  dst[i] = (n < Nsrc) ? (bf16_t)src[(size_t)k * Nsrc + n] : (bf16_t)0.0f;
}

// x[l,b,:] = concat(word_table[inp_word[b,l]], pos_table[inp_pos[b,l]]) as bf16
__global__ void embed_kernel(const int* __restrict__ iw, const int* __restrict__ ip,
                             const float* __restrict__ wt, const float* __restrict__ pt,
                             bf16_t* __restrict__ X) {
  int t = blockIdx.x;          // t = l*32 + b
  int l = t >> 5, b = t & 31;
  int d = threadIdx.x;         // 0..191
  int wid = iw[b * L_SZ + l];
  int pid = ip[b * L_SZ + l];
  float v = (d < 128) ? wt[(size_t)wid * 128 + d] : pt[pid * 64 + (d - 128)];
  X[(size_t)t * D_SZ + d] = (bf16_t)v;
}

// ---------------------------------------------------------------------------
// Input projection: P = X[16384,192] @ W[192,1024] + bias, output stored in
// C-fragment-major bf16: P[(mt*64 + nt)*32 + lane][r], identical (lane,r)
// mapping to the WMMA accumulator so the LSTM consumes it with one b128 load.
// Block = 16 rows x 1024 cols (8 waves x 8 n-tiles). A tile staged via async.
// ---------------------------------------------------------------------------

__global__ __launch_bounds__(256) void preact_gemm_kernel(const bf16_t* __restrict__ X,
                                                          const bf16_t* __restrict__ Wp,
                                                          const float* __restrict__ bias,
                                                          bf16_t* __restrict__ P) {
  __shared__ bf16_t xtile[16 * D_SZ];  // 6 KB
  int tid = threadIdx.x, lane = tid & 31, w = tid >> 5;
  int row0 = blockIdx.x * 16;

  // async stage 16x192 contiguous A tile (6144 B = 384 chunks of 16B)
  for (int i = tid; i < 384; i += 256)
    async_copy_b128((char*)xtile + (size_t)i * 16,
                    (const char*)X + (size_t)row0 * D_SZ * 2 + (size_t)i * 16);
  async_join();

  v8f acc[8] = {};
#pragma unroll
  for (int kc = 0; kc < 6; ++kc) {       // K = 192
    v16bf a = load_a_frag(xtile, D_SZ, 0, kc * 32, lane);
#pragma unroll
    for (int i = 0; i < 8; ++i) {
      v16bf b = load_b_packed(Wp, kc, w * 8 + i, lane);
      acc[i] = wmma_bf16(a, b, acc[i]);
    }
  }
#pragma unroll
  for (int i = 0; i < 8; ++i) {
    int nt = w * 8 + i;
    float bv = bias[nt * 16 + (lane & 15)];
    v8bf pk;
#pragma unroll
    for (int r = 0; r < 8; ++r) pk[r] = (bf16_t)(acc[i][r] + bv);
    *(v8bf*)(P + (((size_t)blockIdx.x * 64 + nt) * 32 + lane) * 8) = pk;
  }
}

// ---------------------------------------------------------------------------
// Persistent BiLSTM. grid = 2 (fwd/bwd), block = 1024 (32 waves).
// Wave (mtile = w&1, j = w>>1) owns gate tiles {j, 16+j, 32+j, 48+j} so the
// i/f/g/o pre-activations and the c/h state for its (b,h) elements stay in
// registers. h matrix (16 KB bf16) lives in LDS; WMMA A-frags read it back.
// preact read as one b128 fragment per gate (fragment-major layout).
// ---------------------------------------------------------------------------

__global__ __launch_bounds__(1024) void bilstm_kernel(const bf16_t* __restrict__ preF,
                                                      const bf16_t* __restrict__ preB,
                                                      const bf16_t* __restrict__ WhF,
                                                      const bf16_t* __restrict__ WhB,
                                                      const float* __restrict__ masks,
                                                      bf16_t* __restrict__ ctx) {
  __shared__ bf16_t hsh[B_SZ * H_SZ];   // 16 KB
  int dir = blockIdx.x;
  const bf16_t* pre = dir ? preB : preF;
  const bf16_t* Wh  = dir ? WhB  : WhF;
  int coff = dir ? H_SZ : 0;

  int tid = threadIdx.x, lane = tid & 31, w = tid >> 5;
  int mtile = w & 1;
  int j     = w >> 1;
  int bbase = mtile * 16 + ((lane >> 4) << 3);
  int hcol  = j * 16 + (lane & 15);

  for (int i = tid; i < B_SZ * H_SZ; i += 1024) hsh[i] = (bf16_t)0.0f;
  v8f cst = {};
  v8f hst = {};
  __syncthreads();

  for (int s = 0; s < L_SZ; ++s) {
    int l = dir ? (L_SZ - 1 - s) : s;
    v8f zi = {}, zf = {}, zg = {}, zo = {};
#pragma unroll
    for (int kc = 0; kc < 8; ++kc) {    // K = 256
      v16bf a  = load_a_frag(hsh, H_SZ, mtile * 16, kc * 32, lane);
      v16bf b0 = load_b_packed(Wh, kc, 0 * 16 + j, lane);
      zi = wmma_bf16(a, b0, zi);
      v16bf b1 = load_b_packed(Wh, kc, 1 * 16 + j, lane);
      zf = wmma_bf16(a, b1, zf);
      v16bf b2 = load_b_packed(Wh, kc, 2 * 16 + j, lane);
      zg = wmma_bf16(a, b2, zg);
      v16bf b3 = load_b_packed(Wh, kc, 3 * 16 + j, lane);
      zo = wmma_bf16(a, b3, zo);
    }
    __syncthreads();                    // all reads of h_prev complete

    int mt = l * 2 + mtile;             // preact row-tile index
    v8bf pi = *(const v8bf*)(pre + (((size_t)mt * 64 + (0 * 16 + j)) * 32 + lane) * 8);
    v8bf pf = *(const v8bf*)(pre + (((size_t)mt * 64 + (1 * 16 + j)) * 32 + lane) * 8);
    v8bf pg = *(const v8bf*)(pre + (((size_t)mt * 64 + (2 * 16 + j)) * 32 + lane) * 8);
    v8bf po = *(const v8bf*)(pre + (((size_t)mt * 64 + (3 * 16 + j)) * 32 + lane) * 8);
#pragma unroll
    for (int r = 0; r < 8; ++r) {
      int b = bbase + r;
      float vi = zi[r] + (float)pi[r];
      float vf = zf[r] + (float)pf[r];
      float vg = zg[r] + (float)pg[r];
      float vo = zo[r] + (float)po[r];
      float ig = 1.0f / (1.0f + expf(-vi));
      float fg = 1.0f / (1.0f + expf(-vf));
      float gg = tanhf(vg);
      float og = 1.0f / (1.0f + expf(-vo));
      float cn = fg * cst[r] + ig * gg;
      float hn = og * tanhf(cn);
      float mt2 = masks[b * L_SZ + l];
      cst[r] = mt2 * cn + (1.0f - mt2) * cst[r];
      hst[r] = mt2 * hn + (1.0f - mt2) * hst[r];
      hsh[b * H_SZ + hcol] = (bf16_t)hst[r];
      ctx[((size_t)b * L_SZ + l) * (2 * H_SZ) + coff + hcol] = (bf16_t)hst[r];
    }
    __syncthreads();                    // h updated for next step
  }
}

// ---------------------------------------------------------------------------
// Decoder: logits = ctx[16384,512] @ Wd + b, fused log-softmax + gumbel-max
// sample + argmax + output scatter. Block = 16 token rows; A tile (16 KB,
// contiguous) staged to LDS via async copies and shared by all 8 waves.
// ---------------------------------------------------------------------------

__device__ inline unsigned mix_hash(unsigned x) {
  x ^= x >> 17; x *= 0xed5ad4bbu;
  x ^= x >> 11; x *= 0xac4c1b51u;
  x ^= x >> 15; x *= 0x31848babu;
  x ^= x >> 14;
  return x;
}

__global__ __launch_bounds__(256) void decoder_kernel(const bf16_t* __restrict__ ctx,
                                                      const bf16_t* __restrict__ Wd,
                                                      const float* __restrict__ bd,
                                                      const int* __restrict__ inp_pos,
                                                      const int* __restrict__ inp_word,
                                                      const int* __restrict__ tgtwords,
                                                      const int* __restrict__ lut,
                                                      float* __restrict__ out) {
  __shared__ bf16_t atile[16 * 512];   // 16 KB
  __shared__ float lg[16 * NTP];       // 64 KB logits tile
  __shared__ float red[256];
  __shared__ int   redi[256];
  __shared__ float smax[16];
  __shared__ float slz[16];
  __shared__ int   ssmp[16];
  __shared__ int   sctc[16];
  __shared__ float spsmp[16];

  int tid = threadIdx.x, lane = tid & 31, w = tid >> 5;
  int row0 = blockIdx.x * 16;

  // async stage 16x512 contiguous A tile (16384 B = 1024 chunks of 16B)
  for (int i = tid; i < 1024; i += 256)
    async_copy_b128((char*)atile + (size_t)i * 16,
                    (const char*)ctx + (size_t)row0 * 512 * 2 + (size_t)i * 16);
  async_join();

  // ---- GEMM phase ----
  v8f acc[8] = {};
#pragma unroll
  for (int kc = 0; kc < 16; ++kc) {    // K = 512
    v16bf a = load_a_frag(atile, 512, 0, kc * 32, lane);
#pragma unroll
    for (int i = 0; i < 8; ++i) {
      v16bf b = load_b_packed(Wd, kc, w * 8 + i, lane);
      acc[i] = wmma_bf16(a, b, acc[i]);
    }
  }
#pragma unroll
  for (int i = 0; i < 8; ++i) {
    int n = (w * 8 + i) * 16 + (lane & 15);
#pragma unroll
    for (int r = 0; r < 8; ++r) {
      int m = ((lane >> 4) << 3) + r;
      lg[m * NTP + n] = (n < NTGT) ? (acc[i][r] + bd[n]) : -1.0e30f;
    }
  }
  __syncthreads();

  // ---- softmax / sample phase: 16 threads per row ----
  int row = tid >> 4, sub = tid & 15;

  float mx = -1.0e30f;
  for (int c = sub; c < NTGT; c += 16) mx = fmaxf(mx, lg[row * NTP + c]);
  red[tid] = mx;
  __syncthreads();
  if (sub == 0) {
    float m2 = -1.0e30f;
    for (int i = 0; i < 16; ++i) m2 = fmaxf(m2, red[row * 16 + i]);
    smax[row] = m2;
  }
  __syncthreads();

  float sm = 0.0f;
  for (int c = sub; c < NTGT; c += 16) sm += expf(lg[row * NTP + c] - smax[row]);
  red[tid] = sm;
  __syncthreads();
  if (sub == 0) {
    float s2 = 0.0f;
    for (int i = 0; i < 16; ++i) s2 += red[row * 16 + i];
    slz[row] = logf(s2);
  }
  __syncthreads();

  int trow = row0 + row;
  float bg = -1.0e30f; int bgi = 0;
  float bv = -1.0e30f; int bvi = 0;
  for (int c = sub; c < NTGT; c += 16) {
    float ls = lg[row * NTP + c] - smax[row] - slz[row];
    float d  = logf(expf(ls) + 1e-12f);             // log(softmax + 1e-12)
    unsigned hx = mix_hash((unsigned)(trow * NTP + c) * 2654435761u + 42u);
    float u = fminf((float)(hx >> 8) * (1.0f / 16777216.0f) + 1e-12f, 0.99999f);
    float g = -logf(-logf(u));                      // gumbel noise
    float dg = d + g;
    if (dg > bg) { bg = dg; bgi = c; }
    if (d  > bv) { bv = d;  bvi = c; }
  }
  red[tid] = bg; redi[tid] = bgi;
  __syncthreads();
  if (sub == 0) {
    float best = -1.0e30f; int bi = 0;
    for (int i = 0; i < 16; ++i)
      if (red[row * 16 + i] > best) { best = red[row * 16 + i]; bi = redi[row * 16 + i]; }
    ssmp[row] = bi;
    float ls = lg[row * NTP + bi] - smax[row] - slz[row];
    spsmp[row] = logf(expf(ls) + 1e-12f);
  }
  __syncthreads();
  red[tid] = bv; redi[tid] = bvi;
  __syncthreads();
  if (sub == 0) {
    float best = -1.0e30f; int bi = 0;
    for (int i = 0; i < 16; ++i)
      if (red[row * 16 + i] > best) { best = red[row * 16 + i]; bi = redi[row * 16 + i]; }
    sctc[row] = bi;
  }
  __syncthreads();

  // ---- output scatter ----
  int t = trow;
  int b = t >> 9, l = t & (L_SZ - 1);
  bool sel = inp_pos[b * L_SZ + l] < 10;            // XPOS_THRESH
  int iw = inp_word[b * L_SZ + l];
  int ow = sel ? tgtwords[ssmp[row]] : iw;
  int cw = sel ? tgtwords[sctc[row]] : iw;

  const size_t O_OW = 0;           // obf_word  [16384]
  const size_t O_OC = 16384;       // obf_char  [16384*16]
  const size_t O_CW = 278528;      // ctc_word  [16384]
  const size_t O_CC = 294912;      // ctc_char  [16384*16]
  const size_t O_PL = 557056;      // plog      [16384]
  const size_t O_OM = 573440;      // obf_mask  [16384]

  out[O_OC + (size_t)t * CHL + sub] = (float)lut[ow * CHL + sub];
  out[O_CC + (size_t)t * CHL + sub] = (float)lut[cw * CHL + sub];
  if (sub == 0) {
    out[O_OW + t] = (float)ow;
    out[O_CW + t] = (float)cw;
    out[O_PL + t] = sel ? spsmp[row] : 0.0f;
    out[O_OM + t] = sel ? 1.0f : 0.0f;
  }
}

// ---------------------------------------------------------------------------
// Host launcher
// ---------------------------------------------------------------------------

extern "C" void kernel_launch(void* const* d_in, const int* in_sizes, int n_in,
                              void* d_out, int out_size, void* d_ws, size_t ws_size,
                              hipStream_t stream) {
  const int*   inp_word   = (const int*)d_in[0];
  const int*   inp_pos    = (const int*)d_in[1];
  const float* masks      = (const float*)d_in[2];
  const float* word_table = (const float*)d_in[3];
  const float* pos_table  = (const float*)d_in[4];
  const float* Wx_f       = (const float*)d_in[5];
  const float* Wh_f       = (const float*)d_in[6];
  const float* b_f        = (const float*)d_in[7];
  const float* Wx_b       = (const float*)d_in[8];
  const float* Wh_b       = (const float*)d_in[9];
  const float* b_b        = (const float*)d_in[10];
  const float* dec_W      = (const float*)d_in[11];
  const float* dec_b      = (const float*)d_in[12];
  const int*   tgtwords   = (const int*)d_in[13];
  const int*   lut        = (const int*)d_in[14];
  float* out = (float*)d_out;

  char* ws = (char*)d_ws;
  size_t off = 0;
  auto alloc = [&](size_t bytes) -> void* {
    void* p = ws + off;
    off += (bytes + 255) & ~(size_t)255;
    return p;
  };
  bf16_t* x_bf   = (bf16_t*)alloc((size_t)16384 * D_SZ * 2);   // 6.3 MB
  bf16_t* wxf    = (bf16_t*)alloc((size_t)D_SZ * G4 * 2);      // packed
  bf16_t* wxb    = (bf16_t*)alloc((size_t)D_SZ * G4 * 2);
  bf16_t* whf    = (bf16_t*)alloc((size_t)H_SZ * G4 * 2);
  bf16_t* whb    = (bf16_t*)alloc((size_t)H_SZ * G4 * 2);
  bf16_t* wdd    = (bf16_t*)alloc((size_t)512 * NTP * 2);
  bf16_t* preF   = (bf16_t*)alloc((size_t)16384 * G4 * 2);     // frag-major
  bf16_t* preB   = (bf16_t*)alloc((size_t)16384 * G4 * 2);
  bf16_t* ctx_bf = (bf16_t*)alloc((size_t)16384 * 512 * 2);
  (void)ws_size; (void)in_sizes; (void)n_in; (void)out_size;

  pack_b_kernel<<<(D_SZ * G4 + 255) / 256, 256, 0, stream>>>(Wx_f, wxf, D_SZ * G4, G4);
  pack_b_kernel<<<(D_SZ * G4 + 255) / 256, 256, 0, stream>>>(Wx_b, wxb, D_SZ * G4, G4);
  pack_b_kernel<<<(H_SZ * G4 + 255) / 256, 256, 0, stream>>>(Wh_f, whf, H_SZ * G4, G4);
  pack_b_kernel<<<(H_SZ * G4 + 255) / 256, 256, 0, stream>>>(Wh_b, whb, H_SZ * G4, G4);
  pack_b_kernel<<<(512 * NTP + 255) / 256, 256, 0, stream>>>(dec_W, wdd, 512 * NTP, NTGT);
  embed_kernel<<<16384, D_SZ, 0, stream>>>(inp_word, inp_pos, word_table, pos_table, x_bf);

  preact_gemm_kernel<<<1024, 256, 0, stream>>>(x_bf, wxf, b_f, preF);
  preact_gemm_kernel<<<1024, 256, 0, stream>>>(x_bf, wxb, b_b, preB);

  bilstm_kernel<<<2, 1024, 0, stream>>>(preF, preB, whf, whb, masks, ctx_bf);

  decoder_kernel<<<1024, 256, 0, stream>>>(ctx_bf, wdd, dec_b, inp_pos, inp_word,
                                           tgtwords, lut, out);
}
